// ButterflyMlp_36112085025100
// MI455X (gfx1250) — matrix operand order repaired and checked by reading, not compiled, not run on hardware
//
#include <hip/hip_runtime.h>
#include <math.h>

// ---------------------------------------------------------------------------
// ButterflyMlp fused forward for gfx1250 (MI455X), bf16 WMMA path.
// 2x2 register-blocked WMMA macro-tiles: 4 independent accumulator chains,
// halved A(LDS)/B(L2) operand traffic per v_wmma.
// ---------------------------------------------------------------------------

typedef __attribute__((ext_vector_type(16))) __bf16 v16bf;
typedef __attribute__((ext_vector_type(8)))  float  v8f;

union TileU { uint4 u[2]; v16bf v; };

__device__ __forceinline__ unsigned short f2bf(float f) {
  union { float f; unsigned int u; } c; c.f = f;
  unsigned int u = c.u;
  return (unsigned short)((u + 0x7FFFu + ((u >> 16) & 1u)) >> 16); // RNE
}

// Python-equivalent butterfly mask (band half-width 10, 5 stripes of width 3).
__device__ __forceinline__ bool bmask(int i, int j, int out_f, int in_f) {
  int jc = (i * in_f) / out_f;
  int d  = j - jc;
  int ad = d < 0 ? -d : d;
  if (ad <= 10) return true;
  int period = in_f / 5; if (period < 1) period = 1;
  int m = d % period; if (m < 0) m += period;   // Python modulo
  return m < 3;
}

#define K1P  800   // 784 padded to multiple of 32 (K of layers 1/2)
#define N1P  800   // layer-1 output rows padded to multiple of 32
#define LD1  808   // LDS row stride for 800-col tiles (bank-conflict pad)
#define LD2  136   // LDS row stride for 128-col tile  (bank-conflict pad)
#define MT   64    // batch rows per workgroup

// ---------------------------------------------------------------------------
// Prep: masked weights -> bf16, K-padded (and N-padded for W1), in workspace.
// ---------------------------------------------------------------------------
__global__ __launch_bounds__(256)
void bf_prep(const float* __restrict__ W1, const float* __restrict__ W2,
             const float* __restrict__ W3,
             unsigned short* __restrict__ W1p, unsigned short* __restrict__ W2p,
             unsigned short* __restrict__ W3p) {
  int idx = blockIdx.x * 256 + threadIdx.x;
  const int n1 = N1P * K1P, n2 = 128 * K1P, n3 = 16 * 128;
  if (idx < n1) {
    int n = idx / K1P, k = idx % K1P;
    float v = 0.f;
    if (n < 784 && k < 784 && bmask(n, k, 784, 784)) v = W1[n * 784 + k];
    W1p[idx] = f2bf(v);
  } else if (idx < n1 + n2) {
    int t = idx - n1; int n = t / K1P, k = t % K1P;
    float v = 0.f;
    if (k < 784 && bmask(n, k, 128, 784)) v = W2[n * 784 + k];
    W2p[t] = f2bf(v);
  } else if (idx < n1 + n2 + n3) {
    int t = idx - n1 - n2; int n = t / 128, k = t % 128;
    float v = 0.f;
    if (n < 10 && bmask(n, k, 10, 128)) v = W3[n * 128 + k];
    W3p[t] = f2bf(v);
  }
}

// ---------------------------------------------------------------------------
// WMMA operand loaders (row-major bf16 source, source = LDS or global).
// A (16x32): lane l holds row (l&15); K = {kh..kh+7, kh+16..kh+23}, kh=8*(l>>4)
// B (32x16) = W^T: lane l holds W-row n0+(l&15); K = {kh..kh+7, kh+16..kh+23}, kh=8*(l>>4)
// ---------------------------------------------------------------------------
__device__ __forceinline__ v16bf load_a(const unsigned short* base, int ld,
                                        int row0, int k0, int lane) {
  int r  = row0 + (lane & 15);
  int kh = (lane >> 4) << 3;
  const unsigned short* p = base + r * ld + k0 + kh;
  TileU t;
  t.u[0] = *(const uint4*)(p);
  t.u[1] = *(const uint4*)(p + 16);
  return t.v;
}
__device__ __forceinline__ v16bf load_b(const unsigned short* base, int ld,
                                        int n0, int k0, int lane) {
  int n  = n0 + (lane & 15);
  int kh = (lane >> 4) << 3;
  const unsigned short* p = base + n * ld + k0 + kh;
  TileU t;
  t.u[0] = *(const uint4*)(p);
  t.u[1] = *(const uint4*)(p + 16);
  return t.v;
}

#define WMMA_BF16(a, b, c) \
  __builtin_amdgcn_wmma_f32_16x16x32_bf16(false, (a), false, (b), (short)0, (c), false, false)

__device__ __forceinline__ void store_relu_tile(unsigned short* dst, int ld,
                                                int rbase, int ncol, float bias,
                                                const v8f& c) {
#pragma unroll
  for (int v = 0; v < 8; ++v) {
    float val = c[v] + bias;
    val = val > 0.f ? val : 0.f;
    dst[(rbase + v) * ld + ncol] = f2bf(val);
  }
}

// ---------------------------------------------------------------------------
// Fused MLP: per workgroup, 64 batch rows end-to-end.
// LDS: xs[64][808] bf16, h1s[64][808] bf16, h2s[64][136] bf16, lg[64][16] f32
// ---------------------------------------------------------------------------
__global__ __launch_bounds__(256)
void bf_mlp_main(const float* __restrict__ x,
                 const float* __restrict__ b1, const float* __restrict__ b2,
                 const float* __restrict__ b3,
                 const unsigned short* __restrict__ W1p,
                 const unsigned short* __restrict__ W2p,
                 const unsigned short* __restrict__ W3p,
                 float* __restrict__ out, int Btot) {
  extern __shared__ char smem[];
  unsigned short* xs  = (unsigned short*)smem;        // [64][LD1]
  unsigned short* h1s = xs  + MT * LD1;               // [64][LD1]
  unsigned short* h2s = h1s + MT * LD1;               // [64][LD2]
  float*          lg  = (float*)(h2s + MT * LD2);     // [64][16]

  const int tid  = threadIdx.x;
  const int lane = tid & 31;
  const int wave = tid >> 5;
  const int r0   = blockIdx.x * MT;

  const v8f zero = {0.f, 0.f, 0.f, 0.f, 0.f, 0.f, 0.f, 0.f};

  // ---- stage x (fp32 -> bf16) into LDS, zero K-pad ----
  for (int i = tid; i < MT * K1P; i += 256) {
    int r = i / K1P, k = i - r * K1P;
    float v = 0.f;
    if (k < 784 && (r0 + r) < Btot) v = x[(size_t)(r0 + r) * 784 + k];
    xs[r * LD1 + k] = f2bf(v);
  }
  __syncthreads();

  // ---- layer 1: h1 = relu(x @ W1m^T + b1)   [64 x 800 (784 real)] ----
  // jobs: 2 M-pairs x 25 N-pairs, each job = 32x32 output macro-tile.
  for (int job = wave; job < 2 * 25; job += 8) {
    int mp = job / 25, np = job % 25;
    int m0 = mp * 32, n0 = np * 32;
    v8f c00 = zero, c01 = zero, c10 = zero, c11 = zero;
    for (int ks = 0; ks < K1P / 32; ++ks) {
      v16bf a0 = load_a(xs,  LD1, m0,      ks * 32, lane);
      v16bf a1 = load_a(xs,  LD1, m0 + 16, ks * 32, lane);
      v16bf bb0 = load_b(W1p, K1P, n0,      ks * 32, lane);
      v16bf bb1 = load_b(W1p, K1P, n0 + 16, ks * 32, lane);
      c00 = WMMA_BF16(a0, bb0, c00);
      c01 = WMMA_BF16(a0, bb1, c01);
      c10 = WMMA_BF16(a1, bb0, c10);
      c11 = WMMA_BF16(a1, bb1, c11);
    }
    int   nc0 = n0 + (lane & 15), nc1 = nc0 + 16;
    float bias0 = (nc0 < 784) ? b1[nc0] : 0.f;   // padded N rows -> h1 pad = 0
    float bias1 = (nc1 < 784) ? b1[nc1] : 0.f;
    int   rb0 = m0 + ((lane >> 4) << 3), rb1 = rb0 + 16;
    store_relu_tile(h1s, LD1, rb0, nc0, bias0, c00);
    store_relu_tile(h1s, LD1, rb0, nc1, bias1, c01);
    store_relu_tile(h1s, LD1, rb1, nc0, bias0, c10);
    store_relu_tile(h1s, LD1, rb1, nc1, bias1, c11);
  }
  __syncthreads();

  // ---- layer 2: h2 = relu(h1 @ W2m^T + b2)  [64 x 128] ----
  // jobs: 2 M-pairs x 4 N-pairs = 8 (one per wave).
  for (int job = wave; job < 2 * 4; job += 8) {
    int mp = job >> 2, np = job & 3;
    int m0 = mp * 32, n0 = np * 32;
    v8f c00 = zero, c01 = zero, c10 = zero, c11 = zero;
    for (int ks = 0; ks < K1P / 32; ++ks) {
      v16bf a0 = load_a(h1s, LD1, m0,      ks * 32, lane);
      v16bf a1 = load_a(h1s, LD1, m0 + 16, ks * 32, lane);
      v16bf bb0 = load_b(W2p, K1P, n0,      ks * 32, lane);
      v16bf bb1 = load_b(W2p, K1P, n0 + 16, ks * 32, lane);
      c00 = WMMA_BF16(a0, bb0, c00);
      c01 = WMMA_BF16(a0, bb1, c01);
      c10 = WMMA_BF16(a1, bb0, c10);
      c11 = WMMA_BF16(a1, bb1, c11);
    }
    int   nc0 = n0 + (lane & 15), nc1 = nc0 + 16;
    float bias0 = b2[nc0], bias1 = b2[nc1];
    int   rb0 = m0 + ((lane >> 4) << 3), rb1 = rb0 + 16;
    store_relu_tile(h2s, LD2, rb0, nc0, bias0, c00);
    store_relu_tile(h2s, LD2, rb0, nc1, bias1, c01);
    store_relu_tile(h2s, LD2, rb1, nc0, bias0, c10);
    store_relu_tile(h2s, LD2, rb1, nc1, bias1, c11);
  }
  __syncthreads();

  // ---- layer 3: logits = h2 @ W3m^T + b3    [64 x 10 (padded 16)] ----
  for (int job = wave; job < 2; job += 8) {   // waves 0..1, paired M
    int m0 = job * 32;
    v8f c0 = zero, c1 = zero;
    for (int ks = 0; ks < 4; ++ks) {
      v16bf a0 = load_a(h2s, LD2, m0,      ks * 32, lane);
      v16bf a1 = load_a(h2s, LD2, m0 + 16, ks * 32, lane);
      v16bf bb = load_b(W3p, 128, 0,       ks * 32, lane);
      c0 = WMMA_BF16(a0, bb, c0);
      c1 = WMMA_BF16(a1, bb, c1);
    }
    int   ncol = lane & 15;
    float bias = (ncol < 10) ? b3[ncol] : 0.f;
    int   rb0 = m0 + ((lane >> 4) << 3), rb1 = rb0 + 16;
#pragma unroll
    for (int v = 0; v < 8; ++v) {
      lg[(rb0 + v) * 16 + ncol] = c0[v] + bias;
      lg[(rb1 + v) * 16 + ncol] = c1[v] + bias;
    }
  }
  __syncthreads();

  // ---- log_softmax over the 10 real columns, write out ----
  for (int r = tid; r < MT; r += 256) {
    if ((r0 + r) >= Btot) continue;
    const float* row = lg + r * 16;
    float mx = row[0];
#pragma unroll
    for (int j = 1; j < 10; ++j) mx = row[j] > mx ? row[j] : mx;
    float s = 0.f;
#pragma unroll
    for (int j = 0; j < 10; ++j) s += expf(row[j] - mx);
    float lse = mx + logf(s);
    float* o = out + (size_t)(r0 + r) * 10;
#pragma unroll
    for (int j = 0; j < 10; ++j) o[j] = row[j] - lse;
  }
}

// ---------------------------------------------------------------------------
// Launch
// ---------------------------------------------------------------------------
extern "C" void kernel_launch(void* const* d_in, const int* in_sizes, int n_in,
                              void* d_out, int out_size, void* d_ws, size_t ws_size,
                              hipStream_t stream) {
  const float* x  = (const float*)d_in[0];
  const float* W1 = (const float*)d_in[1];
  const float* b1 = (const float*)d_in[2];
  const float* W2 = (const float*)d_in[3];
  const float* b2 = (const float*)d_in[4];
  const float* W3 = (const float*)d_in[5];
  const float* b3 = (const float*)d_in[6];
  float* out = (float*)d_out;

  const int Btot = in_sizes[0] / 784;

  char* ws = (char*)d_ws;
  unsigned short* W1p = (unsigned short*)ws;                              // 800*800
  unsigned short* W2p = (unsigned short*)(ws + (size_t)N1P * K1P * 2);    // 128*800
  unsigned short* W3p = (unsigned short*)(ws + (size_t)N1P * K1P * 2
                                             + (size_t)128 * K1P * 2);    // 16*128

  const int prepN = N1P * K1P + 128 * K1P + 16 * 128;
  bf_prep<<<(prepN + 255) / 256, 256, 0, stream>>>(W1, W2, W3, W1p, W2p, W3p);

  const size_t ldsBytes = (size_t)MT * LD1 * 2 * 2   // xs + h1s
                        + (size_t)MT * LD2 * 2       // h2s
                        + (size_t)MT * 16 * 4;       // logits
  hipFuncSetAttribute(reinterpret_cast<const void*>(bf_mlp_main),
                      hipFuncAttributeMaxDynamicSharedMemorySize, (int)ldsBytes);

  const int nWG = (Btot + MT - 1) / MT;
  bf_mlp_main<<<nWG, 256, ldsBytes, stream>>>(x, b1, b2, b3, W1p, W2p, W3p,
                                              out, Btot);
}
